// LiDAREncoder_59674275610938
// MI455X (gfx1250) — matrix operand order repaired and checked by reading, not compiled, not run on hardware
//
#include <hip/hip_runtime.h>
#include <math.h>

typedef __attribute__((ext_vector_type(16))) _Float16 v16h;
typedef __attribute__((ext_vector_type(8)))  _Float16 v8h;
typedef __attribute__((ext_vector_type(8)))  float    v8f;

#define B_   16
#define N_   2048
#define C1_  64
#define N1_  1639          // ceil(0.8*2048)
#define C2_  128
#define N2_  820           // ceil(0.5*1639)
#define KNB  8
#define RT_  ((N1_ + 15) / 16)   // 103 row/col tiles for stage-2 kNN

// ---------- helpers ----------
__device__ __forceinline__ void top8_insert(float d, int j, float* bd, int* bi) {
  if (d < bd[7]) {
    int p = 7;
    while (p > 0 && bd[p-1] > d) { bd[p] = bd[p-1]; bi[p] = bi[p-1]; --p; }
    bd[p] = d; bi[p] = j;
  }
}

// ---------- kernel 1: scalar prep (pw norms, W2 -> W2^T in f16) ----------
__global__ void prep_kernel(const float* pw1, const float* pw2, const float* w2,
                            _Float16* w2t, float* scal) {
  int tid = threadIdx.x;
  if (tid == 0) {
    float s = 0.f;
    for (int k = 0; k < C1_; ++k) s += pw1[k] * pw1[k];
    scal[0] = rsqrtf(s);
    s = 0.f;
    for (int k = 0; k < C2_; ++k) s += pw2[k] * pw2[k];
    scal[1] = rsqrtf(s);
  }
  for (int e = tid; e < C2_ * C2_; e += blockDim.x) {
    int k = e >> 7, n = e & 127;
    w2t[n * C2_ + k] = (_Float16)w2[e];      // transposed: row = output col
  }
}

// ---------- kernel 2: stage-1 kNN in 1-D ----------
__global__ __launch_bounds__(256) void knn1_kernel(const float* x, int* idx1) {
  __shared__ float xs[N_];
  int b = blockIdx.x >> 3;
  int chunk = blockIdx.x & 7;
  const float* xb = x + (size_t)b * N_;
  for (int i = threadIdx.x; i < N_; i += 256) xs[i] = xb[i];
  __syncthreads();
  int p = chunk * 256 + threadIdx.x;
  float xi = xs[p];
  float bd[KNB]; int bi[KNB];
  for (int t = 0; t < KNB; ++t) { bd[t] = INFINITY; bi[t] = 0; }
  for (int j = 0; j < N_; ++j) {
    if (j == p) continue;
    float dx = xi - xs[j];
    top8_insert(dx * dx, j, bd, bi);
  }
  for (int t = 0; t < KNB; ++t) idx1[((size_t)b * N_ + p) * KNB + t] = bi[t];
}

// ---------- kernel 3: stage-1 EdgeConv (2->64) + mean + score1 ----------
__global__ __launch_bounds__(256) void edgeconv1_kernel(
    const float* x, const int* idx1, const float* w1, const float* b1,
    const float* pw1, const float* scal, float* h1, float* score1) {
  int b = blockIdx.x >> 3;
  int p = (blockIdx.x & 7) * 256 + threadIdx.x;
  const float* xb = x + (size_t)b * N_;
  float xi = xb[p];
  float acc[C1_];
  #pragma unroll
  for (int k = 0; k < C1_; ++k) acc[k] = 0.f;
  for (int t = 0; t < KNB; ++t) {
    int j = idx1[((size_t)b * N_ + p) * KNB + t];
    float dv = xb[j] - xi;   // feat = [xi, xj - xi]
    #pragma unroll
    for (int k = 0; k < C1_; ++k)
      acc[k] += fmaxf(xi * w1[k] + dv * w1[C1_ + k] + b1[k], 0.f);
  }
  float dot = 0.f;
  float* outp = h1 + ((size_t)b * N_ + p) * C1_;
  #pragma unroll
  for (int k = 0; k < C1_; ++k) {
    acc[k] *= 0.125f;              // mean over 8 neighbors (ReLU'd -> outer relu is id)
    dot += acc[k] * pw1[k];
    outp[k] = acc[k];
  }
  score1[(size_t)b * N_ + p] = tanhf(dot * scal[0]);
}

// ---------- kernel 4: stage-1 top-k pool: bitonic sort + scaled gather ----------
__global__ __launch_bounds__(256) void sort1_kernel(
    const float* h1, const float* score1, _Float16* h1ph, float* norm1) {
  __shared__ float key[N_];
  __shared__ int   sid[N_];
  int b = blockIdx.x;
  for (int i = threadIdx.x; i < N_; i += 256) {
    key[i] = score1[(size_t)b * N_ + i];
    sid[i] = i;
  }
  __syncthreads();
  for (int k = 2; k <= N_; k <<= 1)
    for (int j = k >> 1; j > 0; j >>= 1) {
      for (int t = threadIdx.x; t < N_ / 2; t += 256) {
        int i = 2 * t - (t & (j - 1));
        int p = i + j;
        bool desc = ((i & k) == 0);
        float ki = key[i], kp = key[p];
        if ((ki < kp) == desc) {
          key[i] = kp; key[p] = ki;
          int s = sid[i]; sid[i] = sid[p]; sid[p] = s;
        }
      }
      __syncthreads();
    }
  for (int r = threadIdx.x; r < N1_; r += 256) {
    int sel = sid[r]; float s = key[r];
    const float* src = h1 + ((size_t)b * N_ + sel) * C1_;
    _Float16* dst = h1ph + ((size_t)b * N1_ + r) * C1_;
    float nrm = 0.f;
    for (int k = 0; k < C1_; ++k) {
      float v = src[k] * s;                    // x_sel * s_sel
      nrm += v * v;
      dst[k] = (_Float16)v;
    }
    norm1[(size_t)b * N1_ + r] = nrm;
  }
}

// ---------- kernel 5: stage-2 kNN via WMMA gram matrix ----------
__global__ __launch_bounds__(32) void knn2_kernel(
    const _Float16* h1ph, const float* norm1, int* idx2) {
  int b  = blockIdx.x / RT_;
  int rt = blockIdx.x % RT_;
  int lane = threadIdx.x;
  int g = lane >> 4, ln = lane & 15;
  const _Float16* Hb = h1ph + (size_t)b * N1_ * C1_;
  const float* nb = norm1 + (size_t)b * N1_;

  // A fragments (16 rows x 64 K, two K-chunks of 32).
  // k order used here (B matches): lane L, half s -> K = s + 16*(L>=16)
  int arow = rt * 16 + ln; if (arow > N1_ - 1) arow = N1_ - 1;
  const v8h* abase = (const v8h*)(Hb + (size_t)arow * C1_);
  v16h a0, a1;
  {
    v8h lo = abase[2 * g], hi = abase[2 * g + 1];
    #pragma unroll
    for (int i = 0; i < 8; ++i) { a0[i] = lo[i]; a0[i + 8] = hi[i]; }
    lo = abase[4 + 2 * g]; hi = abase[5 + 2 * g];
    #pragma unroll
    for (int i = 0; i < 8; ++i) { a1[i] = lo[i]; a1[i + 8] = hi[i]; }
  }
  float nq[8];
  #pragma unroll
  for (int v = 0; v < 8; ++v) {
    int r = rt * 16 + v + 8 * g; if (r > N1_ - 1) r = N1_ - 1;
    nq[v] = nb[r];
  }
  float bd[KNB]; int bi[KNB];
  for (int t = 0; t < KNB; ++t) { bd[t] = INFINITY; bi[t] = 0; }
  __shared__ float tile[16][17];
  int myrow = rt * 16 + lane;   // only meaningful for lanes 0..15

  for (int ct = 0; ct < RT_; ++ct) {
    // B fragment: lane holds column n = ct*16+ln; K = s + 16g (+32*kc) -> contiguous 32B
    int crow = ct * 16 + ln; if (crow > N1_ - 1) crow = N1_ - 1;
    const v16h* cb = (const v16h*)(Hb + (size_t)crow * C1_);
    v16h b0 = cb[g], b1v = cb[2 + g];
    float ncv = nb[crow];
    v8f acc = {};
    acc = __builtin_amdgcn_wmma_f32_16x16x32_f16(false, a0, false, b0,  (short)0, acc, false, false);
    acc = __builtin_amdgcn_wmma_f32_16x16x32_f16(false, a1, false, b1v, (short)0, acc, false, false);
    #pragma unroll
    for (int v = 0; v < 8; ++v) {
      int m  = v + 8 * g;                 // C layout: M = v + 8*(lane>=16)
      int rg = rt * 16 + m;
      int cg = ct * 16 + ln;
      float d = nq[v] + ncv - 2.0f * acc[v];
      if (cg >= N1_ || cg == rg) d = INFINITY;   // mask OOB cols + self
      tile[m][ln] = d;
    }
    __syncthreads();
    if (lane < 16 && myrow < N1_) {
      #pragma unroll
      for (int c = 0; c < 16; ++c) top8_insert(tile[lane][c], ct * 16 + c, bd, bi);
    }
    __syncthreads();
  }
  if (lane < 16 && myrow < N1_) {
    for (int t = 0; t < KNB; ++t) idx2[((size_t)b * N1_ + myrow) * KNB + t] = bi[t];
  }
}

// ---------- kernel 6: stage-2 EdgeConv (128->128) via WMMA, mean fused ----------
__global__ __launch_bounds__(32) void edgeconv2_kernel(
    const _Float16* h1ph, const int* idx2, const _Float16* w2t,
    const float* b2, float* h2) {
  int b  = blockIdx.x / N2_;
  int pw = blockIdx.x % N2_;
  int p0 = pw * 2;                       // 2 points x 8 neighbors = 16 A-rows
  int lane = threadIdx.x;
  int g = lane >> 4, ln = lane & 15;
  const _Float16* Hb = h1ph + (size_t)b * N1_ * C1_;
  __shared__ _Float16 feat[16][C2_];
  for (int e = lane; e < 16 * C2_; e += 32) {
    int r = e >> 7, k = e & 127;
    int p = p0 + (r >> 3); if (p > N1_ - 1) p = N1_ - 1;
    if (k < C1_) {
      feat[r][k] = Hb[(size_t)p * C1_ + k];                       // xi
    } else {
      int j  = idx2[((size_t)b * N1_ + p) * KNB + (r & 7)];
      int kk = k - C1_;
      feat[r][k] = (_Float16)((float)Hb[(size_t)j * C1_ + kk] -
                              (float)Hb[(size_t)p * C1_ + kk]);   // xj - xi
    }
  }
  __syncthreads();
  v16h A[4];
  const v8h* fb = (const v8h*)&feat[ln][0];
  #pragma unroll
  for (int kc = 0; kc < 4; ++kc) {
    v8h lo = fb[2 * g + 4 * kc], hi = fb[2 * g + 1 + 4 * kc];
    #pragma unroll
    for (int i = 0; i < 8; ++i) { A[kc][i] = lo[i]; A[kc][i + 8] = hi[i]; }
  }
  int myp = p0 + g;
  for (int t = 0; t < 8; ++t) {
    int col = t * 16 + ln;
    const v16h* wb = (const v16h*)(w2t + (size_t)col * C2_);  // W2^T row = out col
    v8f acc = {};
    #pragma unroll
    for (int kc = 0; kc < 4; ++kc) {
      v16h bf = wb[g + 2 * kc];
      acc = __builtin_amdgcn_wmma_f32_16x16x32_f16(false, A[kc], false, bf,
                                                   (short)0, acc, false, false);
    }
    float bias = b2[col];
    float s = 0.f;
    #pragma unroll
    for (int v = 0; v < 8; ++v) s += fmaxf(acc[v] + bias, 0.f);  // ReLU then sum
    s *= 0.125f;                                                 // mean over 8 nbrs
    if (myp < N1_) h2[((size_t)b * N1_ + myp) * C2_ + col] = s;  // lanes<16 -> p0, >=16 -> p1
  }
}

// ---------- kernel 7: stage-2 pool (score + sort + weighted mean) ----------
__global__ __launch_bounds__(256) void pool2_kernel(
    const float* h2, const float* pw2, const float* scal, float* out) {
  __shared__ float key[N_];
  __shared__ int   sid[N_];
  __shared__ float part[2][C2_];
  int b = blockIdx.x;
  float inv = scal[1];
  for (int r = threadIdx.x; r < N_; r += 256) {
    float s;
    if (r < N1_) {
      const float* row = h2 + ((size_t)b * N1_ + r) * C2_;
      float d = 0.f;
      for (int k = 0; k < C2_; ++k) d += row[k] * pw2[k];
      s = tanhf(d * inv);
    } else s = -INFINITY;      // pad
    key[r] = s; sid[r] = r;
  }
  __syncthreads();
  for (int k = 2; k <= N_; k <<= 1)
    for (int j = k >> 1; j > 0; j >>= 1) {
      for (int t = threadIdx.x; t < N_ / 2; t += 256) {
        int i = 2 * t - (t & (j - 1));
        int p = i + j;
        bool desc = ((i & k) == 0);
        float ki = key[i], kp = key[p];
        if ((ki < kp) == desc) {
          key[i] = kp; key[p] = ki;
          int s2 = sid[i]; sid[i] = sid[p]; sid[p] = s2;
        }
      }
      __syncthreads();
    }
  int c = threadIdx.x & 127, h = threadIdx.x >> 7;
  float s = 0.f;
  for (int r = h; r < N2_; r += 2)
    s += h2[((size_t)b * N1_ + sid[r]) * C2_ + c] * key[r];   // x_sel * s_sel
  part[h][c] = s;
  __syncthreads();
  if (threadIdx.x < C2_)
    out[(size_t)b * C2_ + threadIdx.x] =
        (part[0][threadIdx.x] + part[1][threadIdx.x]) * (1.0f / N2_);
}

// ---------- host launcher ----------
extern "C" void kernel_launch(void* const* d_in, const int* in_sizes, int n_in,
                              void* d_out, int out_size, void* d_ws, size_t ws_size,
                              hipStream_t stream) {
  (void)in_sizes; (void)n_in; (void)out_size; (void)ws_size;
  const float* x   = (const float*)d_in[0];
  const float* w1  = (const float*)d_in[1];
  const float* b1  = (const float*)d_in[2];
  const float* pw1 = (const float*)d_in[3];
  const float* w2  = (const float*)d_in[4];
  const float* b2  = (const float*)d_in[5];
  const float* pw2 = (const float*)d_in[6];
  float* out = (float*)d_out;

  char* w = (char*)d_ws;
  size_t off = 0;
  auto take = [&](size_t bytes) -> char* {
    char* p = w + off;
    off = (off + bytes + 255) & ~(size_t)255;
    return p;
  };
  int*      idx1   = (int*)     take((size_t)B_ * N_  * KNB * 4);
  float*    h1     = (float*)   take((size_t)B_ * N_  * C1_ * 4);
  float*    score1 = (float*)   take((size_t)B_ * N_  * 4);
  _Float16* h1ph   = (_Float16*)take((size_t)B_ * N1_ * C1_ * 2);
  float*    norm1  = (float*)   take((size_t)B_ * N1_ * 4);
  int*      idx2   = (int*)     take((size_t)B_ * N1_ * KNB * 4);
  float*    h2     = (float*)   take((size_t)B_ * N1_ * C2_ * 4);
  _Float16* w2t    = (_Float16*)take((size_t)C2_ * C2_ * 2);
  float*    scal   = (float*)   take(256);

  prep_kernel     <<<1,        256, 0, stream>>>(pw1, pw2, w2, w2t, scal);
  knn1_kernel     <<<B_ * 8,   256, 0, stream>>>(x, idx1);
  edgeconv1_kernel<<<B_ * 8,   256, 0, stream>>>(x, idx1, w1, b1, pw1, scal, h1, score1);
  sort1_kernel    <<<B_,       256, 0, stream>>>(h1, score1, h1ph, norm1);
  knn2_kernel     <<<B_ * RT_,  32, 0, stream>>>(h1ph, norm1, idx2);
  edgeconv2_kernel<<<B_ * N2_,  32, 0, stream>>>(h1ph, idx2, w2t, b2, h2);
  pool2_kernel    <<<B_,       256, 0, stream>>>(h2, pw2, scal, out);
}
